// TrittentionCube_29978871726703
// MI455X (gfx1250) — matrix operand order repaired and checked by reading, not compiled, not run on hardware
//
#include <hip/hip_runtime.h>
#include <hip/hip_bf16.h>

// ---------------------------------------------------------------------------
// TrittentionCube for MI455X (gfx1250), wave32 + WMMA bf16 path.
//
// Pipeline:
//   pack: f32 -> bf16 operand layouts (A row-major, B stored as BT[col][k])
//   proj: a,b,c,d,e = x @ W_* + bias            (WMMA, K=512)
//   bsum: sum_n b[:, n, :]                       (elementwise reduce)
//   step1 = c_flat @ W_K^T                       (WMMA, 384x4096, K=512)
//   step2 = step1 @ bsum^T   (stored transposed) (WMMA, K=64)
//   u     = d @ W_V (per head, stored as uT[k][p*64+g]) (WMMA, K=64)
//   attn  : per (b,n,r): scores=a@step2 (WMMA) -> masked -> global softmax
//           (single max/denom over 192x192+extra slot, all in LDS) ->
//           w = P @ e (WMMA) -> w written bf16 to scratch
//   z     : Z[16r x 64k] = W[16 x 12288] @ uT   (WMMA, K=12288, prefetched)
//   out   = z @ W_O^T + b_O                      (WMMA, K=512)
// ---------------------------------------------------------------------------

typedef __bf16 bf16;
typedef __attribute__((ext_vector_type(16))) __bf16 v16bf;
typedef __attribute__((ext_vector_type(8)))  float  v8f;

// ---------------- WMMA fragment helpers (wave32, 16x16x32 bf16) -------------
// A (16x32 bf16): lane l -> row m=l&15, k-base kb=(l>>4)*8; vector elems
//   0..7 = A[m, kb+0..7], 8..15 = A[m, kb+16..23]  (two b128 loads)
static __device__ __forceinline__ v16bf load_frag_a(const bf16* A, int lda, int lane) {
  const bf16* p = A + (lane & 15) * lda + ((lane >> 4) << 3);
  v16bf f;
  ((uint4*)&f)[0] = *(const uint4*)(p);
  ((uint4*)&f)[1] = *(const uint4*)(p + 16);
  return f;
}
// B (32x16 bf16) loaded from BT[col][k] (transposed storage): lane l ->
//   col n=l&15, kb=(l>>4)*8; elems 0..7 = BT[n, kb+0..7], 8..15 = BT[n, kb+16..23] (two b128 loads)
static __device__ __forceinline__ v16bf load_frag_bt(const bf16* BT, int ldbt, int lane) {
  const bf16* p = BT + (lane & 15) * ldbt + ((lane >> 4) << 3);
  v16bf f;
  ((uint4*)&f)[0] = *(const uint4*)(p);
  ((uint4*)&f)[1] = *(const uint4*)(p + 16);
  return f;
}
// C/D f32 16x16: elem v -> row = v + 8*(lane>=16), col = lane&15
static __device__ __forceinline__ v8f gemm_tile(const bf16* A, int lda,
                                                const bf16* BT, int ldbt,
                                                int ktiles, int lane) {
  v8f c = {0.f,0.f,0.f,0.f,0.f,0.f,0.f,0.f};
  for (int kt = 0; kt < ktiles; ++kt) {
    v16bf av = load_frag_a(A + kt * 32, lda, lane);
    v16bf bv = load_frag_bt(BT + kt * 32, ldbt, lane);
    c = __builtin_amdgcn_wmma_f32_16x16x32_bf16(false, av, false, bv,
                                                (short)0, c, false, false);
  }
  return c;
}

// ---------------- pack kernels ---------------------------------------------
__global__ void k_pack_x(const float* x, bf16* xbf) {
  int i = blockIdx.x * 256 + threadIdx.x;
  if (i < 384 * 512) xbf[i] = (bf16)x[i];
}
// WT[w][col*512 + d] = W_w[n, d, h], col = n*64+h  (BT layout for projections)
__global__ void k_pack_wt(const float* WA, const float* WB, const float* WC,
                          const float* WD, const float* WE, bf16* WT) {
  int i = blockIdx.x * 256 + threadIdx.x;
  if (i >= 5 * 262144) return;
  int w = i / 262144, o = i % 262144;
  int col = o >> 9, d = o & 511;
  int n = col >> 6, h = col & 63;
  const float* W = (w == 0) ? WA : (w == 1) ? WB : (w == 2) ? WC : (w == 3) ? WD : WE;
  WT[i] = (bf16)W[n * 32768 + d * 64 + h];
}
// WKT[ij*512 + nk] = W_K[n, i, j, k]   (BT: col=ij, k=nk)
__global__ void k_pack_wk(const float* WK, bf16* WKT) {
  int idx = blockIdx.x * 256 + threadIdx.x;
  if (idx >= 2097152) return;
  int ij = idx >> 9, nk = idx & 511;
  int i = ij >> 6, j = ij & 63, n = nk >> 6, k = nk & 63;
  WKT[idx] = (bf16)WK[n * 262144 + i * 4096 + j * 64 + k];
}
// WVT[n][gk*64 + h] = W_V[n, h, g, k]  (BT: col=gk, k=h)
__global__ void k_pack_wv(const float* WV, bf16* WVT) {
  int idx = blockIdx.x * 256 + threadIdx.x;
  if (idx >= 2097152) return;
  int n = idx >> 18, o = idx & 262143;
  int gk = o >> 6, h = o & 63;
  int g = gk >> 6, k = gk & 63;
  WVT[idx] = (bf16)WV[n * 262144 + h * 4096 + g * 64 + k];
}
// WOT[d*512 + nh] = W_O[nh, d]  (BT: col=d, k=nh)
__global__ void k_pack_wo(const float* WO, bf16* WOT) {
  int i = blockIdx.x * 256 + threadIdx.x;
  if (i >= 262144) return;
  int d = i >> 9, nh = i & 511;
  WOT[i] = (bf16)WO[nh * 512 + d];
}

// ---------------- projections ----------------------------------------------
__global__ void __launch_bounds__(256)
k_proj(const bf16* xbf, const bf16* WT,
       const float* bA, const float* bB, const float* bC, const float* bD, const float* bE,
       bf16* a_bf, float* b_raw, bf16* c_bf, bf16* d_bf, bf16* eT_bf) {
  int lane = threadIdx.x & 31;
  int w = blockIdx.x * 8 + (threadIdx.x >> 5);
  if (w >= 5 * 768) return;
  int widx = w / 768, t = w % 768;
  int m0 = (t / 32) * 16, n0 = (t % 32) * 16;
  v8f c = gemm_tile(xbf + m0 * 512, 512, WT + widx * 262144 + n0 * 512, 512, 16, lane);
  const float* bias = (widx == 0) ? bA : (widx == 1) ? bB : (widx == 2) ? bC
                    : (widx == 3) ? bD : bE;
#pragma unroll
  for (int v = 0; v < 8; ++v) {
    int m = m0 + v + ((lane >> 4) << 3);     // b*192 + p
    int col = n0 + (lane & 15);              // n*64 + h
    float val = c[v] + bias[col];
    int bb = m / 192, p = m % 192, n = col >> 6, h = col & 63;
    if (widx == 0)      a_bf[((bb * 8 + n) * 192 + p) * 64 + h] = (bf16)val;
    else if (widx == 1) b_raw[m * 512 + col] = val;
    else if (widx == 2) c_bf[m * 512 + col] = (bf16)val;
    else if (widx == 3) d_bf[((bb * 8 + n) * 192 + p) * 64 + h] = (bf16)val;
    else                eT_bf[((bb * 8 + n) * 64 + h) * 192 + p] = (bf16)val;
  }
}

__global__ void k_bsum(const float* b_raw, bf16* bsum) {
  int i = blockIdx.x * 256 + threadIdx.x;
  if (i >= 2 * 192 * 64) return;
  int bq = i >> 6, j = i & 63;
  float s = 0.f;
#pragma unroll
  for (int n = 0; n < 8; ++n) s += b_raw[bq * 512 + n * 64 + j];
  bsum[i] = (bf16)s;
}

// step1[(b,r), i, j] = sum_{nk} c[(b,r), nk] * WKT[ij, nk]
__global__ void __launch_bounds__(256)
k_step1(const bf16* c_bf, const bf16* WKT, bf16* step1) {
  int lane = threadIdx.x & 31;
  int w = blockIdx.x * 8 + (threadIdx.x >> 5);
  if (w >= 24 * 256) return;
  int m0 = (w / 256) * 16, n0 = (w % 256) * 16;
  v8f c = gemm_tile(c_bf + m0 * 512, 512, WKT + (size_t)n0 * 512, 512, 16, lane);
#pragma unroll
  for (int v = 0; v < 8; ++v) {
    int m = m0 + v + ((lane >> 4) << 3);
    int col = n0 + (lane & 15);
    int i = col >> 6, j = col & 63;
    step1[((size_t)m * 64 + i) * 64 + j] = (bf16)c[v];
  }
}

// step2T[b][r][q][i] = sum_j step1[b][(r,i)][j] * bsum[b][q][j]
__global__ void __launch_bounds__(256)
k_step2(const bf16* step1, const bf16* bsum, bf16* step2T) {
  int lane = threadIdx.x & 31;
  int w = blockIdx.x * 8 + (threadIdx.x >> 5);
  if (w >= 2 * 768 * 12) return;
  int bb = w / 9216, t = w % 9216;
  int m0 = (t / 12) * 16, q0 = (t % 12) * 16;
  v8f c = gemm_tile(step1 + (size_t)bb * 786432 + m0 * 64, 64,
                    bsum + bb * 12288 + q0 * 64, 64, 2, lane);
#pragma unroll
  for (int v = 0; v < 8; ++v) {
    int m = m0 + v + ((lane >> 4) << 3);   // r*64 + i
    int q = q0 + (lane & 15);
    int r = m >> 6, i = m & 63;
    step2T[(((size_t)bb * 192 + r) * 192 + q) * 64 + i] = (bf16)c[v];
  }
}

// uT[(b,n)][k][p*64+g] = sum_h d[(b,n)][p][h] * W_V[n][h][g][k]
__global__ void __launch_bounds__(256)
k_u(const bf16* d_bf, const bf16* WVT, bf16* uT) {
  int lane = threadIdx.x & 31;
  int w = blockIdx.x * 8 + (threadIdx.x >> 5);
  if (w >= 16 * 3072) return;
  int bn = w / 3072, t = w % 3072;
  int p0 = (t / 256) * 16, gk0 = (t % 256) * 16;
  int n = bn & 7;
  v8f c = gemm_tile(d_bf + (size_t)bn * 12288 + p0 * 64, 64,
                    WVT + (size_t)n * 262144 + gk0 * 64, 64, 2, lane);
#pragma unroll
  for (int v = 0; v < 8; ++v) {
    int p = p0 + v + ((lane >> 4) << 3);
    int gk = gk0 + (lane & 15);
    int g = gk >> 6, k = gk & 63;
    uT[((size_t)bn * 64 + k) * 12288 + p * 64 + g] = (bf16)c[v];
  }
}

// ---------------- fused attention core: one block per (r, n, b) -------------
// Produces w[(b,n)][r][p*64+g] = sum_q softmax(scores)[p,q] * e[q,g]  (bf16)
__global__ void __launch_bounds__(256)
k_attn(const bf16* a_bf, const bf16* step2T, const bf16* eT, bf16* w_ws) {
  extern __shared__ __align__(16) char smem[];
  float* S  = (float*)smem;                          // 192*192 f32
  bf16*  P  = (bf16*)(smem + 147456);                // 192*192 bf16
  float* red = (float*)(smem + 147456 + 73728);      // 256 f32
  const int tid = threadIdx.x, lane = tid & 31, wid = tid >> 5;
  const int r = blockIdx.x, n = blockIdx.y, b = blockIdx.z;

  // Phase 1: scores S[p,q] = (1/64)*sum_i a[p,i]*step2[r,i,q], causal mask
  const bf16* Abase = a_bf + (size_t)((b * 8 + n) * 192) * 64;
  const bf16* Bbase = step2T + (size_t)((b * 192 + r) * 192) * 64;
  for (int t = wid; t < 144; t += 8) {
    int p0 = (t / 12) * 16, q0 = (t % 12) * 16;
    v8f c = gemm_tile(Abase + p0 * 64, 64, Bbase + q0 * 64, 64, 2, lane);
#pragma unroll
    for (int v = 0; v < 8; ++v) {
      int p = p0 + v + ((lane >> 4) << 3);
      int q = q0 + (lane & 15);
      float s = c[v] * (1.0f / 64.0f);
      if (p >= q || p >= r) s = -15625.0f;   // IGNORE / D_HEAD
      S[p * 192 + q] = s;
    }
  }
  __syncthreads();

  // Global softmax over all 192*192 entries + the appended -1000 slot
  float lm = -3.4e38f;
  for (int i = tid; i < 36864; i += 256) lm = fmaxf(lm, S[i]);
  red[tid] = lm; __syncthreads();
  for (int s = 128; s > 0; s >>= 1) {
    if (tid < s) red[tid] = fmaxf(red[tid], red[tid + s]);
    __syncthreads();
  }
  float mx = fmaxf(red[0], -1000.0f);
  __syncthreads();
  float ls = 0.f;
  for (int i = tid; i < 36864; i += 256) ls += __expf(S[i] - mx);
  red[tid] = ls; __syncthreads();
  for (int s = 128; s > 0; s >>= 1) {
    if (tid < s) red[tid] += red[tid + s];
    __syncthreads();
  }
  float inv = 1.0f / (red[0] + __expf(-1000.0f - mx));
  __syncthreads();
  for (int i = tid; i < 36864; i += 256) P[i] = (bf16)(__expf(S[i] - mx) * inv);
  __syncthreads();

  // Phase 2: w[p,g] = sum_q P[p,q] * e[q,g]   (WMMA, A from LDS)
  const bf16* Ebase = eT + (size_t)((b * 8 + n) * 64) * 192;
  bf16* Wout = w_ws + ((size_t)((b * 8 + n) * 192) + r) * 12288;
  for (int t = wid; t < 48; t += 8) {
    int p0 = (t / 4) * 16, g0 = (t % 4) * 16;
    v8f c = gemm_tile(P + p0 * 192, 192, Ebase + g0 * 192, 192, 6, lane);
#pragma unroll
    for (int v = 0; v < 8; ++v) {
      int p = p0 + v + ((lane >> 4) << 3);
      int g = g0 + (lane & 15);
      Wout[p * 64 + g] = (bf16)c[v];
    }
  }
}

// ---------------- z GEMM: Z[r,k] = sum_pg w[r,pg] * uT[k,pg] ----------------
// Per (b,n): 192x64 = [192 x 12288] @ [12288 x 64], K streamed with prefetch.
__global__ void __launch_bounds__(256)
k_z(const bf16* w_ws, const bf16* uT, bf16* z_bf) {
  int lane = threadIdx.x & 31;
  int w = blockIdx.x * 8 + (threadIdx.x >> 5);
  if (w >= 16 * 48) return;
  int bn = w / 48, t = w % 48;
  int r0 = (t / 4) * 16, k0 = (t % 4) * 16;
  const bf16* A  = w_ws + (size_t)bn * 192 * 12288 + (size_t)r0 * 12288
                 + (lane & 15) * 12288 + ((lane >> 4) << 3);
  const bf16* BT = uT + (size_t)bn * 64 * 12288 + (size_t)k0 * 12288
                 + (lane & 15) * 12288 + ((lane >> 4) << 3);
  v8f c = {0.f,0.f,0.f,0.f,0.f,0.f,0.f,0.f};
  for (int kt = 0; kt < 384; ++kt) {
    const bf16* pa = A + kt * 32;
    const bf16* pb = BT + kt * 32;
    if ((kt & 3) == 0) {           // prefetch streaming operands ~512B ahead
      __builtin_prefetch(pa + 256, 0, 1);
      __builtin_prefetch(pb + 256, 0, 1);
    }
    v16bf av, bv;
    ((uint4*)&av)[0] = *(const uint4*)(pa);
    ((uint4*)&av)[1] = *(const uint4*)(pa + 16);
    ((uint4*)&bv)[0] = *(const uint4*)(pb);
    ((uint4*)&bv)[1] = *(const uint4*)(pb + 16);
    c = __builtin_amdgcn_wmma_f32_16x16x32_bf16(false, av, false, bv,
                                                (short)0, c, false, false);
  }
  int b = bn >> 3, n = bn & 7;
#pragma unroll
  for (int v = 0; v < 8; ++v) {
    int r = r0 + v + ((lane >> 4) << 3);
    int k = k0 + (lane & 15);
    z_bf[((size_t)(b * 192 + r)) * 512 + n * 64 + k] = (bf16)c[v];
  }
}

// ---------------- output projection ----------------------------------------
__global__ void __launch_bounds__(256)
k_out(const bf16* z_bf, const bf16* WOT, const float* bO, float* out) {
  int lane = threadIdx.x & 31;
  int w = blockIdx.x * 8 + (threadIdx.x >> 5);
  if (w >= 24 * 32) return;
  int m0 = (w / 32) * 16, n0 = (w % 32) * 16;
  v8f c = gemm_tile(z_bf + m0 * 512, 512, WOT + n0 * 512, 512, 16, lane);
#pragma unroll
  for (int v = 0; v < 8; ++v) {
    int m = m0 + v + ((lane >> 4) << 3);
    int col = n0 + (lane & 15);
    out[m * 512 + col] = c[v] + bO[col];
  }
}

// ---------------- workspace layout (bytes) ---------------------------------
static const size_t OFF_XBF  = 0;         // 384*512       bf16
static const size_t OFF_WT   = 393216;    // 5*512*512     bf16
static const size_t OFF_WKT  = 3014656;   // 4096*512      bf16
static const size_t OFF_WVT  = 7208960;   // 8*4096*64     bf16
static const size_t OFF_WOT  = 11403264;  // 512*512       bf16
static const size_t OFF_ABF  = 11927552;  // 2*8*192*64    bf16
static const size_t OFF_BRAW = 12320768;  // 2*192*512     f32
static const size_t OFF_CBF  = 13107200;  // 384*512       bf16
static const size_t OFF_DBF  = 13500416;  // 2*8*192*64    bf16
static const size_t OFF_ET   = 13893632;  // 2*8*64*192    bf16
static const size_t OFF_BSUM = 14286848;  // 2*192*64      bf16
static const size_t OFF_S1   = 14336000;  // 2*192*64*64   bf16
static const size_t OFF_S2T  = 17481728;  // 2*192*192*64  bf16
static const size_t OFF_UT   = 26918912;  // 16*64*12288   bf16
static const size_t OFF_ZBF  = 52084736;  // 2*192*512     bf16
static const size_t OFF_WWS  = 52477952;  // 16*192*12288  bf16 (75.5 MB)

extern "C" void kernel_launch(void* const* d_in, const int* in_sizes, int n_in,
                              void* d_out, int out_size, void* d_ws, size_t ws_size,
                              hipStream_t stream) {
  (void)in_sizes; (void)n_in; (void)out_size; (void)ws_size;
  const float* x  = (const float*)d_in[0];
  const float* WA = (const float*)d_in[1];
  const float* WB = (const float*)d_in[2];
  const float* WC = (const float*)d_in[3];
  const float* WD = (const float*)d_in[4];
  const float* WE = (const float*)d_in[5];
  const float* WK = (const float*)d_in[6];
  const float* WV = (const float*)d_in[7];
  const float* WO = (const float*)d_in[8];
  const float* bA = (const float*)d_in[9];
  const float* bB = (const float*)d_in[10];
  const float* bC = (const float*)d_in[11];
  const float* bD = (const float*)d_in[12];
  const float* bE = (const float*)d_in[13];
  const float* bO = (const float*)d_in[14];
  float* out = (float*)d_out;

  char* ws = (char*)d_ws;
  bf16* xbf   = (bf16*)(ws + OFF_XBF);
  bf16* WT    = (bf16*)(ws + OFF_WT);
  bf16* WKT   = (bf16*)(ws + OFF_WKT);
  bf16* WVT   = (bf16*)(ws + OFF_WVT);
  bf16* WOT   = (bf16*)(ws + OFF_WOT);
  bf16* a_bf  = (bf16*)(ws + OFF_ABF);
  float* braw = (float*)(ws + OFF_BRAW);
  bf16* c_bf  = (bf16*)(ws + OFF_CBF);
  bf16* d_bf  = (bf16*)(ws + OFF_DBF);
  bf16* eT    = (bf16*)(ws + OFF_ET);
  bf16* bsum  = (bf16*)(ws + OFF_BSUM);
  bf16* s1    = (bf16*)(ws + OFF_S1);
  bf16* s2T   = (bf16*)(ws + OFF_S2T);
  bf16* uT    = (bf16*)(ws + OFF_UT);
  bf16* z_bf  = (bf16*)(ws + OFF_ZBF);
  bf16* w_ws  = (bf16*)(ws + OFF_WWS);

  k_pack_x <<<768,  256, 0, stream>>>(x, xbf);
  k_pack_wt<<<5120, 256, 0, stream>>>(WA, WB, WC, WD, WE, WT);
  k_pack_wk<<<8192, 256, 0, stream>>>(WK, WKT);
  k_pack_wv<<<8192, 256, 0, stream>>>(WV, WVT);
  k_pack_wo<<<1024, 256, 0, stream>>>(WO, WOT);

  k_proj <<<480,  256, 0, stream>>>(xbf, WT, bA, bB, bC, bD, bE,
                                    a_bf, braw, c_bf, d_bf, eT);
  k_bsum <<<96,   256, 0, stream>>>(braw, bsum);
  k_step1<<<768,  256, 0, stream>>>(c_bf, WKT, s1);
  k_step2<<<2304, 256, 0, stream>>>(s1, bsum, s2T);
  k_u    <<<6144, 256, 0, stream>>>(d_bf, WVT, uT);

  dim3 agrid(192, 8, 2);
  size_t smem = 147456 + 73728 + 1024;  // 222208 B < 320 KB/WGP
  k_attn<<<agrid, 256, smem, stream>>>(a_bf, s2T, eT, w_ws);

  k_z  <<<96, 256, 0, stream>>>(w_ws, uT, z_bf);
  k_out<<<96, 256, 0, stream>>>(z_bf, WOT, bO, out);
}
